// JointNet_15625091023608
// MI455X (gfx1250) — matrix operand order repaired and checked by reading, not compiled, not run on hardware
//
#include <hip/hip_runtime.h>
#include <hip/hip_bf16.h>

typedef _Float16 v16h __attribute__((ext_vector_type(16)));
typedef _Float16 v8h  __attribute__((ext_vector_type(8)));
typedef _Float16 v2h  __attribute__((ext_vector_type(2)));
typedef float    v8f  __attribute__((ext_vector_type(8)));

#define WMMA_F32_F16(a, b, c) \
  __builtin_amdgcn_wmma_f32_16x16x32_f16(false, (a), false, (b), (short)0, (c), false, false)

constexpr int Bb = 4, T = 400, U = 80, De = 512, F = 1024, V = 1024;
constexpr int BT = Bb * T;   // 1600 enc rows
constexpr int BU = Bb * U;   // 320 dec rows

// ---------------------------------------------------------------------------
// fp32 -> f16 conversion (grid-stride)
// ---------------------------------------------------------------------------
__global__ void cvt_f32_to_f16(const float* __restrict__ in,
                               _Float16* __restrict__ out, int n) {
  int i = blockIdx.x * blockDim.x + threadIdx.x;
  int stride = gridDim.x * blockDim.x;
  for (; i < n; i += stride) out[i] = (_Float16)in[i];
}

// ---------------------------------------------------------------------------
// Small projection GEMM: out[M][N] = A[M][K] * Bm[N][K]^T + bias[N]
// One wave per 16x16 tile, f16 WMMA with fp32 accumulation.
// M,N multiples of 16; K multiple of 32.
// ---------------------------------------------------------------------------
__global__ __launch_bounds__(256) void proj_gemm_f16(
    const _Float16* __restrict__ A, const _Float16* __restrict__ Bm,
    const float* __restrict__ bias, float* __restrict__ out,
    int M, int N, int K) {
  const int wid = blockIdx.x * (blockDim.x >> 5) + (threadIdx.x >> 5);
  const int lane = threadIdx.x & 31;
  const int ntiles = N >> 4;
  const int total = (M >> 4) * ntiles;
  if (wid >= total) return;               // wave-uniform
  const int mt = wid / ntiles, nt = wid % ntiles;
  const int lh = lane & 15, sel = lane >> 4;

  const _Float16* arow = A + (size_t)(mt * 16 + lh) * K + sel * 16;
  const _Float16* brow = Bm + (size_t)(nt * 16 + lh) * K + sel * 16;

  v8f acc = {0.f, 0.f, 0.f, 0.f, 0.f, 0.f, 0.f, 0.f};
  for (int k0 = 0; k0 < K; k0 += 32) {
    v8h alo = *(const v8h*)(arow + k0);
    v8h ahi = *(const v8h*)(arow + k0 + 8);
    v16h a = __builtin_shufflevector(alo, ahi, 0, 1, 2, 3, 4, 5, 6, 7,
                                     8, 9, 10, 11, 12, 13, 14, 15);
    v16h b = *(const v16h*)(brow + k0);   // 32B contiguous -> 2x b128
    acc = WMMA_F32_F16(a, b, acc);
  }
  const int n = nt * 16 + lh;
  const float bn = bias[n];
#pragma unroll
  for (int g = 0; g < 8; ++g) {
    const int m = mt * 16 + g + sel * 8;
    out[(size_t)m * N + n] = acc[g] + bn;
  }
}

// ---------------------------------------------------------------------------
// Fused joint kernel: one block per (b,t).
//   A tile (LDS, f16): tanh(enc[bt,:] + dec[b,u,:]) for u = 0..79, K=F=1024
//   GEMM vs W_fc (f16, L2-resident), then fused blank/log-softmax epilogue.
// 512 threads = 16 waves; wave w owns output columns [w*64, w*64+64).
// Per-wave accumulators: 5 (M) x 4 (N) WMMA tiles = 160 VGPRs.
// ---------------------------------------------------------------------------
__device__ __forceinline__ float softplus_f(float x) {
  return fmaxf(x, 0.f) + log1pf(__expf(-fabsf(x)));
}

__global__ __launch_bounds__(512, 1) void joint_fused(
    const float* __restrict__ enc,     // [BT][F]
    const float* __restrict__ dec,     // [BU][F]
    const _Float16* __restrict__ Wfc,  // [V][F] row-major, f16
    const float* __restrict__ bfc,     // [V]
    float* __restrict__ out) {         // [BT*U][V]
  constexpr int AS = F + 8;            // padded LDS row stride (halves)
  __shared__ _Float16 Atile[U * AS];   // ~161 KB
  __shared__ float redmax[16][U];
  __shared__ float redsum[16][U];
  __shared__ float rowblank[U];
  __shared__ float rowmax[U];
  __shared__ float rowlse[U];
  __shared__ float logblk[U];
  __shared__ float lognot[U];

  const int bt = blockIdx.x;
  const int b = bt / T;
  const int tid = threadIdx.x;

  // ---- Phase A: build f16 tanh tile in LDS ----
  {
    const int k = tid * 2;  // 512 threads x 2 = F
    const float2 e = *(const float2*)(enc + (size_t)bt * F + k);
    const float* dbase = dec + (size_t)(b * U) * F + k;
    for (int u = 0; u < U; ++u) {
      const float2 dv = *(const float2*)(dbase + (size_t)u * F);
      v2h h;
      h[0] = (_Float16)tanhf(e.x + dv.x);
      h[1] = (_Float16)tanhf(e.y + dv.y);
      *(v2h*)(&Atile[u * AS + k]) = h;
    }
  }
  __syncthreads();

  const int wave = tid >> 5, lane = tid & 31;
  const int lh = lane & 15, sel = lane >> 4;
  const int n0 = wave * 64;

  v8f acc[5][4];
  const v8f vz = {0.f, 0.f, 0.f, 0.f, 0.f, 0.f, 0.f, 0.f};
#pragma unroll
  for (int mi = 0; mi < 5; ++mi)
#pragma unroll
    for (int ni = 0; ni < 4; ++ni) acc[mi][ni] = vz;

  const _Float16* bbase[4];
#pragma unroll
  for (int ni = 0; ni < 4; ++ni)
    bbase[ni] = Wfc + (size_t)(n0 + ni * 16 + lh) * F + sel * 16;
  const _Float16* abase[5];
#pragma unroll
  for (int mi = 0; mi < 5; ++mi)
    abase[mi] = &Atile[(mi * 16 + lh) * AS + sel * 16];

  // ---- Phase B: K-loop, 32 per step, 20 WMMAs per wave per step ----
  for (int k0 = 0; k0 < F; k0 += 32) {
    v16h bfrag[4];
#pragma unroll
    for (int ni = 0; ni < 4; ++ni) bfrag[ni] = *(const v16h*)(bbase[ni] + k0);
#pragma unroll
    for (int mi = 0; mi < 5; ++mi) {
      v8h alo = *(const v8h*)(abase[mi] + k0);        // ds_load_b128
      v8h ahi = *(const v8h*)(abase[mi] + k0 + 8);    // ds_load_b128
      v16h a = __builtin_shufflevector(alo, ahi, 0, 1, 2, 3, 4, 5, 6, 7,
                                       8, 9, 10, 11, 12, 13, 14, 15);
#pragma unroll
      for (int ni = 0; ni < 4; ++ni)
        acc[mi][ni] = WMMA_F32_F16(a, bfrag[ni], acc[mi][ni]);
    }
  }

  // ---- Phase C: fused blank + log-softmax epilogue ----
  float bias_n[4];
#pragma unroll
  for (int ni = 0; ni < 4; ++ni) bias_n[ni] = bfc[n0 + ni * 16 + lh];

  // blank logits (column 0) live in wave 0, ni 0, lanes with lh==0
  if (wave == 0 && lh == 0) {
#pragma unroll
    for (int mi = 0; mi < 5; ++mi)
#pragma unroll
      for (int g = 0; g < 8; ++g)
        rowblank[mi * 16 + g + sel * 8] = acc[mi][0][g] + bias_n[0];
  }

  // per-row max over v=1..1023 (wave-local, then cross-wave)
#pragma unroll
  for (int mi = 0; mi < 5; ++mi) {
#pragma unroll
    for (int g = 0; g < 8; ++g) {
      float vmax = -3.0e38f;
#pragma unroll
      for (int ni = 0; ni < 4; ++ni) {
        const float x = acc[mi][ni][g] + bias_n[ni];
        const bool isblank = (wave == 0) && (ni == 0) && (lh == 0);
        vmax = isblank ? vmax : fmaxf(vmax, x);
      }
#pragma unroll
      for (int off = 8; off >= 1; off >>= 1)
        vmax = fmaxf(vmax, __shfl_xor(vmax, off, 16));
      if (lh == 0) redmax[wave][mi * 16 + g + sel * 8] = vmax;
    }
  }
  __syncthreads();
  if (tid < U) {
    float mx = -3.0e38f;
    for (int w = 0; w < 16; ++w) mx = fmaxf(mx, redmax[w][tid]);
    rowmax[tid] = mx;
  }
  __syncthreads();

  // per-row sum of exp (excluding blank)
#pragma unroll
  for (int mi = 0; mi < 5; ++mi) {
#pragma unroll
    for (int g = 0; g < 8; ++g) {
      const int row = mi * 16 + g + sel * 8;
      const float mx = rowmax[row];
      float s = 0.f;
#pragma unroll
      for (int ni = 0; ni < 4; ++ni) {
        const float x = acc[mi][ni][g] + bias_n[ni];
        const bool isblank = (wave == 0) && (ni == 0) && (lh == 0);
        s += isblank ? 0.f : __expf(x - mx);
      }
#pragma unroll
      for (int off = 8; off >= 1; off >>= 1) s += __shfl_xor(s, off, 16);
      if (lh == 0) redsum[wave][row] = s;
    }
  }
  __syncthreads();
  if (tid < U) {
    float s = 0.f;
    for (int w = 0; w < 16; ++w) s += redsum[w][tid];
    rowlse[tid] = __logf(s) + rowmax[tid];
    const float bl = rowblank[tid];
    logblk[tid] = -softplus_f(-bl);  // log_sigmoid(bl)
    lognot[tid] = -softplus_f(bl);   // log_sigmoid(-bl)
  }
  __syncthreads();

  // final writes (coalesced b32 across lanes)
  float* orow = out + (size_t)bt * U * V;
#pragma unroll
  for (int mi = 0; mi < 5; ++mi) {
#pragma unroll
    for (int ni = 0; ni < 4; ++ni) {
      const int n = n0 + ni * 16 + lh;
#pragma unroll
      for (int g = 0; g < 8; ++g) {
        const int row = mi * 16 + g + sel * 8;
        const float x = acc[mi][ni][g] + bias_n[ni];
        const float ov = (n == 0) ? logblk[row]
                                  : (lognot[row] + x - rowlse[row]);
        orow[(size_t)row * V + n] = ov;
      }
    }
  }
}

// ---------------------------------------------------------------------------
extern "C" void kernel_launch(void* const* d_in, const int* in_sizes, int n_in,
                              void* d_out, int out_size, void* d_ws,
                              size_t ws_size, hipStream_t stream) {
  const float* enc_in = (const float*)d_in[0];
  const float* dec_in = (const float*)d_in[1];
  const float* W_enc = (const float*)d_in[2];
  const float* b_enc = (const float*)d_in[3];
  const float* W_dec = (const float*)d_in[4];
  const float* b_dec = (const float*)d_in[5];
  const float* W_fc = (const float*)d_in[6];
  const float* b_fc = (const float*)d_in[7];
  float* out = (float*)d_out;

  char* ws = (char*)d_ws;
  size_t off = 0;
  auto alloc = [&](size_t bytes) -> void* {
    void* p = ws + off;
    off += (bytes + 255) & ~(size_t)255;
    return p;
  };
  _Float16* enc_h  = (_Float16*)alloc((size_t)BT * De * 2);
  _Float16* dec_h  = (_Float16*)alloc((size_t)BU * De * 2);
  _Float16* Wenc_h = (_Float16*)alloc((size_t)F * De * 2);
  _Float16* Wdec_h = (_Float16*)alloc((size_t)F * De * 2);
  _Float16* Wfc_h  = (_Float16*)alloc((size_t)V * F * 2);
  float* enc_f = (float*)alloc((size_t)BT * F * 4);
  float* dec_f = (float*)alloc((size_t)BU * F * 4);
  (void)ws_size; (void)in_sizes; (void)n_in; (void)out_size;

  auto cvt = [&](const float* src, _Float16* dst, int n) {
    int blocks = (n + 255) / 256;
    if (blocks > 2048) blocks = 2048;
    cvt_f32_to_f16<<<blocks, 256, 0, stream>>>(src, dst, n);
  };
  cvt(enc_in, enc_h, BT * De);
  cvt(dec_in, dec_h, BU * De);
  cvt(W_enc, Wenc_h, F * De);
  cvt(W_dec, Wdec_h, F * De);
  cvt(W_fc, Wfc_h, V * F);

  // enc: M=1600, N=1024, K=512 -> 6400 wave-tiles / 8 waves per block
  proj_gemm_f16<<<(BT / 16) * (F / 16) / 8, 256, 0, stream>>>(
      enc_h, Wenc_h, b_enc, enc_f, BT, F, De);
  // dec: M=320, N=1024, K=512 -> 1280 wave-tiles
  proj_gemm_f16<<<(BU / 16) * (F / 16) / 8, 256, 0, stream>>>(
      dec_h, Wdec_h, b_dec, dec_f, BU, F, De);

  joint_fused<<<BT, 512, 0, stream>>>(enc_f, dec_f, Wfc_h, b_fc, out);
}
